// TransformerBlock_68058051772721
// MI455X (gfx1250) — matrix operand order repaired and checked by reading, not compiled, not run on hardware
//
#include <hip/hip_runtime.h>
#include <hip/hip_bf16.h>

// ---------------------------------------------------------------------------
// Transformer block for MI455X (gfx1250): bf16 WMMA GEMMs + flash attention.
// Pipeline: LN1 -> Q/K/V GEMMs -> flash-attn -> proj(+res) -> LN2
//           -> FFN1(+GELU) -> FFN2(+res)
// Matmuls: V_WMMA_F32_16X16X32_BF16 (fp32 accumulate).
// Tile staging: GLOBAL_LOAD_ASYNC_TO_LDS_B128 with ping-pong LDS double
// buffering (one barrier per K-step; async loads overlap the WMMA pipe).
// ---------------------------------------------------------------------------

#define TOK   8192      // 4 * 2048 tokens
#define SEQ   2048
#define BATCH 4
#define EMB   768
#define HEADS 12
#define HDIM  64
#define FF    3072

typedef __bf16 bf16_t;
typedef bf16_t v16bf __attribute__((ext_vector_type(16)));
typedef float  v8f   __attribute__((ext_vector_type(8)));

union Frag {
    v16bf        v;
    unsigned int u[8];
};

__device__ __forceinline__ unsigned short f2bf(float f) {
    unsigned int u = __builtin_bit_cast(unsigned int, f);
    u += 0x7FFFu + ((u >> 16) & 1u);            // round-to-nearest-even
    return (unsigned short)(u >> 16);
}

__device__ __forceinline__ unsigned int pack2bf(float lo, float hi) {
    return (unsigned int)f2bf(lo) | ((unsigned int)f2bf(hi) << 16);
}

// A-matrix (16x32 bf16) K index for VGPR v, lane-half h  (ISA 7.12.2 table)
__device__ __forceinline__ int kA(int v, int h) {
    return ((v & 4) << 2) + h * 8 + ((v & 3) << 1);
}

__device__ __forceinline__ v8f vzero8() {
    v8f z;
#pragma unroll
    for (int i = 0; i < 8; ++i) z[i] = 0.0f;
    return z;
}

__device__ __forceinline__ v8f wmma_bf16(const Frag& a, const Frag& b, v8f c) {
#if defined(__gfx1250__)
    return __builtin_amdgcn_wmma_f32_16x16x32_bf16(
        /*neg_a=*/false, a.v, /*neg_b=*/false, b.v,
        /*c_mod=*/(short)0, c, /*reuse_a=*/false, /*reuse_b=*/false);
#else
    (void)a; (void)b;
    return c;   // host pass / non-gfx1250: never executed
#endif
}

// Async 16B global -> LDS copy (per-lane), tracked by ASYNCcnt.
// LDS generic pointer low 32 bits == LDS byte offset (ISA 10.2 aperture calc).
__device__ __forceinline__ void async_b128(void* lds_dst, const void* gsrc) {
#if defined(__gfx1250__)
    unsigned int l = (unsigned int)(unsigned long long)lds_dst;
    unsigned long long g = (unsigned long long)gsrc;
    asm volatile("global_load_async_to_lds_b128 %0, %1, off"
                 :: "v"(l), "v"(g) : "memory");
#else
    *reinterpret_cast<uint4*>(lds_dst) =
        *reinterpret_cast<const uint4*>(gsrc);   // host stub, never run
#endif
}

__device__ __forceinline__ void async_wait0() {
#if defined(__gfx1250__)
    asm volatile("s_wait_asynccnt 0x0" ::: "memory");
#endif
}

__device__ __forceinline__ float gelu_tanh(float x) {
    const float c = 0.7978845608028654f;        // sqrt(2/pi)
    float t = tanhf(c * (x + 0.044715f * x * x * x));
    return 0.5f * x * (1.0f + t);
}

// ---------------------------------------------------------------------------
// LayerNorm: fp32 [row,768] -> bf16 [row,768]. One block per row.
// ---------------------------------------------------------------------------
__global__ __launch_bounds__(256) void ln_kernel(
    const float* __restrict__ x, const float* __restrict__ g,
    const float* __restrict__ b, unsigned short* __restrict__ out)
{
    __shared__ float red[256];
    const int row = blockIdx.x;
    const int t   = threadIdx.x;
    const float* xr = x + (size_t)row * EMB;

    float a0 = xr[t], a1 = xr[t + 256], a2 = xr[t + 512];

    red[t] = a0 + a1 + a2;
    __syncthreads();
    for (int o = 128; o > 0; o >>= 1) {
        if (t < o) red[t] += red[t + o];
        __syncthreads();
    }
    const float mean = red[0] * (1.0f / EMB);
    __syncthreads();

    float d0 = a0 - mean, d1 = a1 - mean, d2 = a2 - mean;
    red[t] = d0 * d0 + d1 * d1 + d2 * d2;
    __syncthreads();
    for (int o = 128; o > 0; o >>= 1) {
        if (t < o) red[t] += red[t + o];
        __syncthreads();
    }
    const float rstd = rsqrtf(red[0] * (1.0f / EMB) + 1e-5f);

    unsigned short* orow = out + (size_t)row * EMB;
    orow[t]       = f2bf(g[t]       * d0 * rstd + b[t]);
    orow[t + 256] = f2bf(g[t + 256] * d1 * rstd + b[t + 256]);
    orow[t + 512] = f2bf(g[t + 512] * d2 * rstd + b[t + 512]);
}

// ---------------------------------------------------------------------------
// Generic WMMA GEMM:  C[M,N] = epilogue(A_bf16[M,K] @ B_f32[K,N])
// Block tile 128x128x32, 256 threads = 8 waves (4 M-waves x 2 N-waves),
// each wave computes 32x64 via 2x4 accumulators -> 8 wmma / K-step.
// Ping-pong double buffering: A tile via async global->LDS B128 issued one
// step ahead; B weights (f32) pipelined through registers, converted to
// bf16 packed pairs and stored transposed [n][k]. One barrier per K-step.
// ---------------------------------------------------------------------------
#define LAS 40   // LDS stride (bf16 elems) for A [m][k]; 80B rows, 16B aligned
#define LBS 40   // LDS stride for B^T [n][k]

template <bool OUT_BF16, bool BIAS, bool GELU_, bool RES>
__global__ __launch_bounds__(256) void gemm_bf16_kernel(
    const unsigned short* __restrict__ A,   // bf16 [M,K]
    const float* __restrict__ Bw,           // f32  [K,N]
    const float* __restrict__ bias,         // f32  [N]    (if BIAS)
    const float* __restrict__ res,          // f32  [M,N]  (if RES)
    float* __restrict__ outF,               // f32  [M,N]  (if !OUT_BF16)
    unsigned short* __restrict__ outH,      // bf16 [M,N]  (if OUT_BF16)
    int M, int N, int K)
{
    __shared__ unsigned short lA[2][128 * LAS];
    __shared__ unsigned short lB[2][128 * LBS];

    const int tid  = threadIdx.x;
    const int lane = tid & 31;
    const int wave = tid >> 5;
    const int wm   = wave >> 1;          // 0..3
    const int wn   = wave & 1;           // 0..1
    const int lh   = lane >> 4;          // lane half
    const int ll   = lane & 15;
    const int bm   = blockIdx.y * 128;
    const int bn   = blockIdx.x * 128;

    v8f acc[2][4];
#pragma unroll
    for (int i = 0; i < 2; ++i)
#pragma unroll
        for (int j = 0; j < 4; ++j) acc[i][j] = vzero8();

    // issue async A-tile loads (2 x B128 per thread) for K-step k0
    auto issueA = [&](unsigned short* dst, int k0) {
#pragma unroll
        for (int p = 0; p < 2; ++p) {
            const int idx = p * 2048 + tid * 8;
            const int r = idx >> 5, c = idx & 31;        // c in {0,8,16,24}
            async_b128(&dst[r * LAS + c],
                       A + (size_t)(bm + r) * K + (k0 + c));
        }
    };
    // B pipeline registers: 2 passes x 2 K-rows x float4
    float4 br0[2], br1[2];
    auto gloadB = [&](int k0) {
#pragma unroll
        for (int p = 0; p < 2; ++p) {
            const int slot = p * 256 + tid;
            const int kr = (slot >> 5) * 2;              // even K row
            const int n  = (slot & 31) * 4;
            br0[p] = *reinterpret_cast<const float4*>(
                Bw + (size_t)(k0 + kr) * N + (bn + n));
            br1[p] = *reinterpret_cast<const float4*>(
                Bw + (size_t)(k0 + kr + 1) * N + (bn + n));
        }
    };
    auto storeB = [&](unsigned short* dst) {
#pragma unroll
        for (int p = 0; p < 2; ++p) {
            const int slot = p * 256 + tid;
            const int kr = (slot >> 5) * 2;
            const int n  = (slot & 31) * 4;
            *reinterpret_cast<unsigned int*>(&dst[(n + 0) * LBS + kr]) = pack2bf(br0[p].x, br1[p].x);
            *reinterpret_cast<unsigned int*>(&dst[(n + 1) * LBS + kr]) = pack2bf(br0[p].y, br1[p].y);
            *reinterpret_cast<unsigned int*>(&dst[(n + 2) * LBS + kr]) = pack2bf(br0[p].z, br1[p].z);
            *reinterpret_cast<unsigned int*>(&dst[(n + 3) * LBS + kr]) = pack2bf(br0[p].w, br1[p].w);
        }
    };

    const int nk = K >> 5;
    gloadB(0);
    issueA(lA[0], 0);

    for (int it = 0; it < nk; ++it) {
        const int cur = it & 1;
        storeB(lB[cur]);                       // convert pipelined B regs
        if (it + 1 < nk) gloadB((it + 1) * 32);  // prefetch next B into regs
        async_wait0();                         // this tile's A asyncs done
        __syncthreads();                       // publish lA[cur], lB[cur];
                                               // all waves done with buf^1
        if (it + 1 < nk) issueA(lA[cur ^ 1], (it + 1) * 32);  // overlap wmma

        const unsigned short* pA = lA[cur];
        const unsigned short* pB = lB[cur];
        Frag a[2], b[4];
#pragma unroll
        for (int i = 0; i < 2; ++i) {
            const int row = wm * 32 + i * 16 + ll;
#pragma unroll
            for (int v = 0; v < 8; ++v)
                a[i].u[v] = *reinterpret_cast<const unsigned int*>(
                    &pA[row * LAS + kA(v, lh)]);
        }
#pragma unroll
        for (int j = 0; j < 4; ++j) {
            const int col = wn * 64 + j * 16 + ll;
#pragma unroll
            for (int v = 0; v < 8; ++v)
                b[j].u[v] = *reinterpret_cast<const unsigned int*>(
                    &pB[col * LBS + kA(v, lh)]);
        }
#pragma unroll
        for (int i = 0; i < 2; ++i)
#pragma unroll
            for (int j = 0; j < 4; ++j)
                acc[i][j] = wmma_bf16(a[i], b[j], acc[i][j]);
        // no trailing barrier: next iteration's top barrier provides WAR
    }

    // ---- epilogue ----
#pragma unroll
    for (int i = 0; i < 2; ++i) {
#pragma unroll
        for (int j = 0; j < 4; ++j) {
            const int ncol = bn + wn * 64 + j * 16 + ll;
            float bv = 0.0f;
            if constexpr (BIAS) bv = bias[ncol];
#pragma unroll
            for (int v = 0; v < 8; ++v) {
                const int m = bm + wm * 32 + i * 16 + v + 8 * lh;
                float x = acc[i][j][v] + bv;
                if constexpr (GELU_) x = gelu_tanh(x);
                if constexpr (RES)   x += res[(size_t)m * N + ncol];
                if constexpr (OUT_BF16)
                    outH[(size_t)m * N + ncol] = f2bf(x);
                else
                    outF[(size_t)m * N + ncol] = x;
            }
        }
    }
}

// ---------------------------------------------------------------------------
// Flash attention (causal). Grid: (BATCH*HEADS, SEQ/64). Block: 128 thr = 4
// waves; wave w owns 16 query rows. 32-key chunks, online softmax, ping-pong
// double-buffered K (async B128) and V (register transpose). Two barriers
// per chunk. Q staged once through the flat K region (both buffers).
// ---------------------------------------------------------------------------
#define LKS 72   // lK stride [key][d] (also Q staging); 144B rows, 16B aligned
#define LVS 40   // lV stride [d][key], 32 keys + 8 pad
#define LPS 40   // lP stride [qrow][key], 32 + 8 pad

__global__ __launch_bounds__(128) void attn_kernel(
    const unsigned short* __restrict__ q,
    const unsigned short* __restrict__ k,
    const unsigned short* __restrict__ v,
    unsigned short* __restrict__ ctx)
{
    __shared__ unsigned short lK[2 * 32 * LKS];      // 2 K-chunk buffers;
                                                     // flat 64 rows = Q stage
    __shared__ unsigned short lV[2][64 * LVS];       // V chunk, transposed
    __shared__ unsigned short lP[4][16 * LPS];       // per-wave P tile 16x32

    const int tid  = threadIdx.x;
    const int lane = tid & 31;
    const int w    = tid >> 5;
    const int lh   = lane >> 4;
    const int ll   = lane & 15;
    const int bb   = blockIdx.x / HEADS;
    const int hh   = blockIdx.x % HEADS;
    const int qy   = blockIdx.y;
    const size_t base = (size_t)bb * SEQ * EMB + hh * HDIM;

    // ---- stage Q tile (64 q-rows x 64 d) into flat lK via async copies ----
#pragma unroll
    for (int p = 0; p < 4; ++p) {
        const int idx = p * 1024 + tid * 8;
        const int r = idx >> 6, d = idx & 63;        // d in {0,8,...,56}
        async_b128(&lK[r * LKS + d],
                   q + base + (size_t)(qy * 64 + r) * EMB + d);
    }
    async_wait0();
    __syncthreads();
    Frag qa[2];
#pragma unroll
    for (int kk = 0; kk < 2; ++kk) {
        const int row = w * 16 + ll;
#pragma unroll
        for (int vv = 0; vv < 8; ++vv)
            qa[kk].u[vv] = *reinterpret_cast<const unsigned int*>(
                &lK[row * LKS + kk * 32 + kA(vv, lh)]);
    }
    __syncthreads();                 // Q frags read; K buffers reusable

    // issue chunk c into buffer buf: K rows async, V transposed via regs
    auto issue_chunk = [&](int c, int buf) {
        const int kb = c * 32;
        unsigned short* dk = &lK[buf * 32 * LKS];
#pragma unroll
        for (int p = 0; p < 2; ++p) {
            const int idx = p * 1024 + tid * 8;
            const int kr = idx >> 6, d = idx & 63;
            async_b128(&dk[kr * LKS + d],
                       k + base + (size_t)(kb + kr) * EMB + d);
        }
        const int kr = (tid >> 3) * 2;               // even key row
        const int d0 = (tid & 7) * 8;
        const uint4 t0 = *reinterpret_cast<const uint4*>(
            v + base + (size_t)(kb + kr) * EMB + d0);
        const uint4 t1 = *reinterpret_cast<const uint4*>(
            v + base + (size_t)(kb + kr + 1) * EMB + d0);
        const unsigned short* s0 = reinterpret_cast<const unsigned short*>(&t0);
        const unsigned short* s1 = reinterpret_cast<const unsigned short*>(&t1);
#pragma unroll
        for (int jj = 0; jj < 8; ++jj) {
            const unsigned int pk =
                (unsigned int)s0[jj] | ((unsigned int)s1[jj] << 16);
            *reinterpret_cast<unsigned int*>(&lV[buf][(d0 + jj) * LVS + kr]) = pk;
        }
    };

    float mi[8], li[8];
    v8f cacc[4];
#pragma unroll
    for (int vv = 0; vv < 8; ++vv) { mi[vv] = -3.0e38f; li[vv] = 0.0f; }
#pragma unroll
    for (int jt = 0; jt < 4; ++jt) cacc[jt] = vzero8();

    const int nchunks = 2 * qy + 2;                  // keys [0, 64*qy+64)
    issue_chunk(0, 0);
    for (int c = 0; c < nchunks; ++c) {
        const int buf = c & 1;
        const int kb  = c * 32;
        async_wait0();
        __syncthreads();             // chunk c published; buf^1 free (WAR)
        if (c + 1 < nchunks) issue_chunk(c + 1, buf ^ 1);   // overlap compute

        const unsigned short* pK = &lK[buf * 32 * LKS];
        const unsigned short* pV = lV[buf];

        // ---- scores: two 16x16 tiles covering 32 keys ----
        float ps[2][8];
#pragma unroll
        for (int t = 0; t < 2; ++t) {
            v8f s = vzero8();
            const int key = t * 16 + ll;
            Frag bk;
#pragma unroll
            for (int kk = 0; kk < 2; ++kk) {
#pragma unroll
                for (int vv = 0; vv < 8; ++vv)
                    bk.u[vv] = *reinterpret_cast<const unsigned int*>(
                        &pK[key * LKS + kk * 32 + kA(vv, lh)]);
                s = wmma_bf16(qa[kk], bk, s);
            }
#pragma unroll
            for (int vv = 0; vv < 8; ++vv) {
                const int kgl = kb + t * 16 + ll;
                const int qgl = qy * 64 + w * 16 + vv + 8 * lh;
                float sv = s[vv] * 0.125f;                 // 1/sqrt(64)
                ps[t][vv] = (kgl <= qgl) ? sv : -3.0e38f;  // causal mask
            }
        }

        // ---- online softmax update (row reduce across 16 lanes) ----
        float fct[8];
#pragma unroll
        for (int vv = 0; vv < 8; ++vv) {
            float x = fmaxf(ps[0][vv], ps[1][vv]);
            x = fmaxf(x, __shfl_xor(x, 1, 32));
            x = fmaxf(x, __shfl_xor(x, 2, 32));
            x = fmaxf(x, __shfl_xor(x, 4, 32));
            x = fmaxf(x, __shfl_xor(x, 8, 32));
            const float mn = fmaxf(mi[vv], x);
            fct[vv] = __expf(mi[vv] - mn);
            mi[vv]  = mn;
            const float p0 = __expf(ps[0][vv] - mn);
            const float p1 = __expf(ps[1][vv] - mn);
            float rs = p0 + p1;
            rs += __shfl_xor(rs, 1, 32);
            rs += __shfl_xor(rs, 2, 32);
            rs += __shfl_xor(rs, 4, 32);
            rs += __shfl_xor(rs, 8, 32);
            li[vv] = li[vv] * fct[vv] + rs;
            const int prow = vv + 8 * lh;
            lP[w][prow * LPS + ll]      = f2bf(p0);
            lP[w][prow * LPS + 16 + ll] = f2bf(p1);
        }
#pragma unroll
        for (int jt = 0; jt < 4; ++jt)
#pragma unroll
            for (int vv = 0; vv < 8; ++vv) cacc[jt][vv] *= fct[vv];
        __syncthreads();             // P visible for all lanes of each wave

        // ---- PV: ctx += P(16x32) @ V(32x64) ----
        Frag pa;
#pragma unroll
        for (int vv = 0; vv < 8; ++vv)
            pa.u[vv] = *reinterpret_cast<const unsigned int*>(
                &lP[w][ll * LPS + kA(vv, lh)]);
#pragma unroll
        for (int jt = 0; jt < 4; ++jt) {
            Frag bv_;
            const int drow = jt * 16 + ll;
#pragma unroll
            for (int vv = 0; vv < 8; ++vv)
                bv_.u[vv] = *reinterpret_cast<const unsigned int*>(
                    &pV[drow * LVS + kA(vv, lh)]);
            cacc[jt] = wmma_bf16(pa, bv_, cacc[jt]);
        }
        // no trailing barrier: next iteration's top barrier provides WAR
    }

    // ---- normalize and write ctx (bf16 [tok, EMB] at head offset) ----
#pragma unroll
    for (int jt = 0; jt < 4; ++jt) {
#pragma unroll
        for (int vv = 0; vv < 8; ++vv) {
            const int qgl = qy * 64 + w * 16 + vv + 8 * lh;
            const int d   = jt * 16 + ll;
            const float val = cacc[jt][vv] / li[vv];
            ctx[base + (size_t)qgl * EMB + d] = f2bf(val);
        }
    }
}

// ---------------------------------------------------------------------------
// Launch. Workspace layout (bytes), total 5*TOK*EMB*2 = 62,914,560:
//   [0]              xn  (bf16 TOK*EMB)  -- reused for LN2 output h
//   [1*TOK*EMB*2]    q,k,v,ctx (bf16, 4x TOK*EMB) -- region reused for h1
// Residual stream x2 lives in d_out (f32 TOK*EMB).
// ---------------------------------------------------------------------------
extern "C" void kernel_launch(void* const* d_in, const int* in_sizes, int n_in,
                              void* d_out, int out_size, void* d_ws, size_t ws_size,
                              hipStream_t stream) {
    (void)in_sizes; (void)n_in; (void)out_size; (void)ws_size;
    const float* x    = (const float*)d_in[0];
    const float* ln1s = (const float*)d_in[1];
    const float* ln1b = (const float*)d_in[2];
    const float* wq   = (const float*)d_in[3];
    const float* wk   = (const float*)d_in[4];
    const float* wv   = (const float*)d_in[5];
    const float* wo   = (const float*)d_in[6];
    const float* bo   = (const float*)d_in[7];
    const float* ln2s = (const float*)d_in[8];
    const float* ln2b = (const float*)d_in[9];
    const float* w1   = (const float*)d_in[10];
    const float* b1   = (const float*)d_in[11];
    const float* w2   = (const float*)d_in[12];
    const float* b2   = (const float*)d_in[13];
    float* out = (float*)d_out;

    const size_t NE = (size_t)TOK * EMB;
    unsigned short* xn = (unsigned short*)d_ws;
    unsigned short* qb = xn + NE;
    unsigned short* kb = qb + NE;
    unsigned short* vb = kb + NE;
    unsigned short* cb = vb + NE;
    unsigned short* h1 = qb;            // reuse q/k/v/ctx region: TOK*FF bf16

    const dim3 blk256(256), blk128(128);
    const dim3 gE(EMB / 128, TOK / 128);     // N=768 GEMMs
    const dim3 gF(FF / 128, TOK / 128);      // N=3072 GEMM

    // 1) LN1
    ln_kernel<<<TOK, blk256, 0, stream>>>(x, ln1s, ln1b, xn);
    // 2) Q, K, V projections (bf16 out)
    gemm_bf16_kernel<true, false, false, false><<<gE, blk256, 0, stream>>>(
        xn, wq, nullptr, nullptr, nullptr, qb, TOK, EMB, EMB);
    gemm_bf16_kernel<true, false, false, false><<<gE, blk256, 0, stream>>>(
        xn, wk, nullptr, nullptr, nullptr, kb, TOK, EMB, EMB);
    gemm_bf16_kernel<true, false, false, false><<<gE, blk256, 0, stream>>>(
        xn, wv, nullptr, nullptr, nullptr, vb, TOK, EMB, EMB);
    // 3) causal flash attention
    attn_kernel<<<dim3(BATCH * HEADS, SEQ / 64), blk128, 0, stream>>>(qb, kb, vb, cb);
    // 4) output projection + bias + residual(x)  -> x2 in d_out (f32)
    gemm_bf16_kernel<false, true, false, true><<<gE, blk256, 0, stream>>>(
        cb, wo, bo, x, out, nullptr, TOK, EMB, EMB);
    // 5) LN2 on x2
    ln_kernel<<<TOK, blk256, 0, stream>>>(out, ln2s, ln2b, xn);
    // 6) FFN1: gelu(h @ w1 + b1) -> h1 (bf16)
    gemm_bf16_kernel<true, true, true, false><<<gF, blk256, 0, stream>>>(
        xn, w1, b1, nullptr, nullptr, h1, TOK, FF, EMB);
    // 7) FFN2: x2 + h1 @ w2 + b2 -> d_out (f32)
    gemm_bf16_kernel<false, true, false, true><<<gE, blk256, 0, stream>>>(
        h1, w2, b2, out, out, nullptr, TOK, EMB, FF);
}
